// StackedAttention_66700842107346
// MI455X (gfx1250) — matrix operand order repaired and checked by reading, not compiled, not run on hardware
//
#include <hip/hip_runtime.h>
#include <hip/hip_bf16.h>

// ---------------------------------------------------------------------------
// StackedAttention for MI455X (gfx1250, wave32, WMMA).
// All GEMMs run as f16-input / f32-accumulate v_wmma_f32_16x16x32_f16.
// ---------------------------------------------------------------------------

typedef __attribute__((ext_vector_type(16))) _Float16 v16h;
typedef __attribute__((ext_vector_type(8)))  float    v8f;

#define WMMA16(A,B,C) __builtin_amdgcn_wmma_f32_16x16x32_f16(false,(A),false,(B),(short)0,(C),false,false)

constexpr int Bb = 256;   // batch
constexpr int Cc = 256;   // channels
constexpr int Nn = 256;   // sequence
constexpr int Ll = 5;     // SA layers

// ---- WMMA fragment loads from LDS (layouts per CDNA5 ISA 7.12.2) ----------
// A 16x32 f16: lanes 0-15 -> row M=lane, K = {kb+0..7, kb+16..23}
//              lanes 16-31 -> same rows, K = {kb+8..15, kb+24..31}
__device__ __forceinline__ v16h ld_afrag(const _Float16* As, int m0, int kb,
                                         int lane, int ld) {
  int row = m0 + (lane & 15);
  int kh  = (lane & 16) ? 8 : 0;
  const _Float16* p = As + row * ld + kb + kh;
  union { uint4 u[2]; v16h h; } f;
  f.u[0] = *(const uint4*)p;        // K = kb+kh+0..7
  f.u[1] = *(const uint4*)(p + 16); // K = kb+kh+16..23
  return f.h;
}
// B 32x16 f16 (stored transposed in LDS as [n][K]):
// lanes 0-15 -> col n=lane, K = {kb+0..7, kb+16..23} ; lanes 16-31 -> K = {kb+8..15, kb+24..31}
__device__ __forceinline__ v16h ld_bfrag(const _Float16* Bst, int n0, int kb,
                                         int lane, int ld) {
  int col = n0 + (lane & 15);
  int ko  = (lane & 16) ? 8 : 0;
  const _Float16* p = Bst + col * ld + kb + ko;
  union { uint4 u[2]; v16h h; } f;
  f.u[0] = *(const uint4*)p;
  f.u[1] = *(const uint4*)(p + 16);
  return f.h;
}

// ---- GEMM core: block = 256 threads = 8 waves, each wave -> 32x32 tile ----
// BM x BN block tile, K chunked by 64 through LDS.
// A panel: if !ATRANS row-major [BM][KK] (stride lda), else [KK][BM] (stride lda).
// B panel: always [KK][BN] (stride ldb), transposed into LDS.
template <int BM, int BN, int KK, bool ATRANS>
__device__ __forceinline__ void gemm_core(const _Float16* __restrict__ Ag, int lda,
                                          const _Float16* __restrict__ Bg, int ldb,
                                          v8f (&acc)[2][2]) {
  constexpr int KC = 64, LD = KC + 8;   // +8 halves pad vs bank conflicts
  __shared__ _Float16 As[BM * LD];
  __shared__ _Float16 Bst[BN * LD];
  const int tid = threadIdx.x, lane = tid & 31, wave = tid >> 5;
  const int wn = wave % (BN / 32);
  const int wm = wave / (BN / 32);

  for (int kc = 0; kc < KK; kc += KC) {
    if (!ATRANS) {
      constexpr int tot = BM * (KC / 8);
      for (int idx = tid; idx < tot; idx += 256) {
        int r = idx >> 3, c = (idx & 7) * 8;
        *(uint4*)&As[r * LD + c] = *(const uint4*)&Ag[(size_t)r * lda + kc + c];
      }
    } else {
      constexpr int tot = KC * (BM / 8);
      for (int idx = tid; idx < tot; idx += 256) {
        int k = idx / (BM / 8), c0 = (idx % (BM / 8)) * 8;
        uint4 u = *(const uint4*)&Ag[(size_t)(kc + k) * lda + c0];
        const _Float16* h = (const _Float16*)&u;
#pragma unroll
        for (int i = 0; i < 8; ++i) As[(c0 + i) * LD + k] = h[i];
      }
    }
    {
      constexpr int tot = KC * (BN / 8);
      for (int idx = tid; idx < tot; idx += 256) {
        int k = idx / (BN / 8), c0 = (idx % (BN / 8)) * 8;
        uint4 u = *(const uint4*)&Bg[(size_t)(kc + k) * ldb + c0];
        const _Float16* h = (const _Float16*)&u;
#pragma unroll
        for (int i = 0; i < 8; ++i) Bst[(c0 + i) * LD + k] = h[i];
      }
    }
    __syncthreads();
#pragma unroll
    for (int ks = 0; ks < KC / 32; ++ks) {
      int kb = ks * 32;
      v16h a0 = ld_afrag(As, wm * 32,      kb, lane, LD);
      v16h a1 = ld_afrag(As, wm * 32 + 16, kb, lane, LD);
      v16h b0 = ld_bfrag(Bst, wn * 32,      kb, lane, LD);
      v16h b1 = ld_bfrag(Bst, wn * 32 + 16, kb, lane, LD);
      acc[0][0] = WMMA16(a0, b0, acc[0][0]);
      acc[0][1] = WMMA16(a0, b1, acc[0][1]);
      acc[1][0] = WMMA16(a1, b0, acc[1][0]);
      acc[1][1] = WMMA16(a1, b1, acc[1][1]);
    }
    __syncthreads();
  }
}

// D layout: VGPR r, lanes 0-15 -> M=r, lanes 16-31 -> M=r+8; N = lane&15.
#define EPI_GM(mb, BMv, i, r, lane) ((mb) * (BMv) + wm * 32 + (i) * 16 + (r) + (((lane) & 16) ? 8 : 0))
#define EPI_GN(nb, BNv, j, lane)    ((nb) * (BNv) + wn * 32 + (j) * 16 + ((lane) & 15))

// ---------------------------------------------------------------------------
// GEMM kernels
// ---------------------------------------------------------------------------

// conv1/conv2: y = relu(scale[o]*(W@X) + bias[o]); writes f32 + f16 mirrors
__global__ __launch_bounds__(256) void k_gemm_convbn(
    const _Float16* __restrict__ W, const _Float16* __restrict__ X,
    const float* __restrict__ scale, const float* __restrict__ bias,
    float* __restrict__ out32, _Float16* __restrict__ out16) {
  const int nb = blockIdx.x, mb = blockIdx.y, b = blockIdx.z;
  const _Float16* Ag = W + (size_t)mb * 128 * Cc;
  const _Float16* Bg = X + (size_t)b * Cc * Nn + nb * 64;
  v8f acc[2][2] = {};
  gemm_core<128, 64, 256, false>(Ag, Cc, Bg, Nn, acc);
  const int lane = threadIdx.x & 31, wave = threadIdx.x >> 5;
  const int wn = wave % 2, wm = wave / 2;
#pragma unroll
  for (int i = 0; i < 2; ++i)
#pragma unroll
    for (int j = 0; j < 2; ++j)
#pragma unroll
      for (int r = 0; r < 8; ++r) {
        int gm = EPI_GM(mb, 128, i, r, lane);
        int gn = EPI_GN(nb, 64, j, lane);
        float v = acc[i][j][r] * scale[gm] + bias[gm];
        v = v > 0.f ? v : 0.f;
        size_t o = (size_t)b * Cc * Nn + (size_t)gm * Nn + gn;
        out32[o] = v;
        out16[o] = (_Float16)v;
      }
}

// xk = wqk @ cur : [B,64,N] f16
__global__ __launch_bounds__(256) void k_gemm_xk(
    const _Float16* __restrict__ W, const _Float16* __restrict__ X,
    _Float16* __restrict__ xk) {
  const int nb = blockIdx.x, b = blockIdx.z;
  const _Float16* Ag = W;  // [64][256]
  const _Float16* Bg = X + (size_t)b * Cc * Nn + nb * 128;
  v8f acc[2][2] = {};
  gemm_core<64, 128, 256, false>(Ag, Cc, Bg, Nn, acc);
  const int lane = threadIdx.x & 31, wave = threadIdx.x >> 5;
  const int wn = wave % 4, wm = wave / 4;
#pragma unroll
  for (int i = 0; i < 2; ++i)
#pragma unroll
    for (int j = 0; j < 2; ++j)
#pragma unroll
      for (int r = 0; r < 8; ++r) {
        int gm = EPI_GM(0, 64, i, r, lane);
        int gn = EPI_GN(nb, 128, j, lane);
        xk[(size_t)b * 64 * Nn + (size_t)gm * Nn + gn] = (_Float16)acc[i][j][r];
      }
}

// xv = wv @ cur + bv : [B,C,N] f16
__global__ __launch_bounds__(256) void k_gemm_xv(
    const _Float16* __restrict__ W, const _Float16* __restrict__ X,
    const float* __restrict__ bv, _Float16* __restrict__ xv) {
  const int nb = blockIdx.x, mb = blockIdx.y, b = blockIdx.z;
  const _Float16* Ag = W + (size_t)mb * 128 * Cc;
  const _Float16* Bg = X + (size_t)b * Cc * Nn + nb * 64;
  v8f acc[2][2] = {};
  gemm_core<128, 64, 256, false>(Ag, Cc, Bg, Nn, acc);
  const int lane = threadIdx.x & 31, wave = threadIdx.x >> 5;
  const int wn = wave % 2, wm = wave / 2;
#pragma unroll
  for (int i = 0; i < 2; ++i)
#pragma unroll
    for (int j = 0; j < 2; ++j)
#pragma unroll
      for (int r = 0; r < 8; ++r) {
        int gm = EPI_GM(mb, 128, i, r, lane);
        int gn = EPI_GN(nb, 64, j, lane);
        xv[(size_t)b * Cc * Nn + (size_t)gm * Nn + gn] =
            (_Float16)(acc[i][j][r] + bv[gm]);
      }
}

// energy[b,n,m] = sum_c xk[b,c,n]*xk[b,c,m]  (A transposed from xk, K=64)
__global__ __launch_bounds__(256) void k_gemm_energy(
    const _Float16* __restrict__ xk, float* __restrict__ eng) {
  const int nb = blockIdx.x, mb = blockIdx.y, b = blockIdx.z;
  const _Float16* base = xk + (size_t)b * 64 * Nn;
  const _Float16* Ag = base + mb * 128;  // [64][128-col slice], transpose-load
  const _Float16* Bg = base + nb * 64;   // [64][64-col slice]
  v8f acc[2][2] = {};
  gemm_core<128, 64, 64, true>(Ag, Nn, Bg, Nn, acc);
  const int lane = threadIdx.x & 31, wave = threadIdx.x >> 5;
  const int wn = wave % 2, wm = wave / 2;
#pragma unroll
  for (int i = 0; i < 2; ++i)
#pragma unroll
    for (int j = 0; j < 2; ++j)
#pragma unroll
      for (int r = 0; r < 8; ++r) {
        int gm = EPI_GM(mb, 128, i, r, lane);  // n index
        int gn = EPI_GN(nb, 64, j, lane);      // m index
        eng[(size_t)b * Nn * Nn + (size_t)gm * Nn + gn] = acc[i][j][r];
      }
}

// x_r raw = xv @ attn : store f32 (consumed only via its row-mean)
__global__ __launch_bounds__(256) void k_gemm_xr(
    const _Float16* __restrict__ xv, const _Float16* __restrict__ attn,
    float* __restrict__ xr) {
  const int nb = blockIdx.x, mb = blockIdx.y, b = blockIdx.z;
  const _Float16* Ag = xv + (size_t)b * Cc * Nn + (size_t)mb * 128 * Nn;
  const _Float16* Bg = attn + (size_t)b * Nn * Nn + nb * 64;
  v8f acc[2][2] = {};
  gemm_core<128, 64, 256, false>(Ag, Nn, Bg, Nn, acc);
  const int lane = threadIdx.x & 31, wave = threadIdx.x >> 5;
  const int wn = wave % 2, wm = wave / 2;
#pragma unroll
  for (int i = 0; i < 2; ++i)
#pragma unroll
    for (int j = 0; j < 2; ++j)
#pragma unroll
      for (int r = 0; r < 8; ++r) {
        int gm = EPI_GM(mb, 128, i, r, lane);
        int gn = EPI_GN(nb, 64, j, lane);
        xr[(size_t)b * Cc * Nn + (size_t)gm * Nn + gn] = acc[i][j][r];
      }
}

// h = relu(scale*(wt@xs) + bias); cur += h (in-place safe: 1:1 element map);
// also writes the layer slice of d_out.
__global__ __launch_bounds__(256) void k_gemm_out(
    const _Float16* __restrict__ W, const _Float16* __restrict__ xs,
    const float* __restrict__ scale, const float* __restrict__ bias,
    float* __restrict__ cur32, _Float16* __restrict__ cur16,
    float* __restrict__ out, int l) {
  const int nb = blockIdx.x, mb = blockIdx.y, b = blockIdx.z;
  const _Float16* Ag = W + (size_t)mb * 128 * Cc;
  const _Float16* Bg = xs + (size_t)b * Cc * Nn + nb * 64;
  v8f acc[2][2] = {};
  gemm_core<128, 64, 256, false>(Ag, Cc, Bg, Nn, acc);
  const int lane = threadIdx.x & 31, wave = threadIdx.x >> 5;
  const int wn = wave % 2, wm = wave / 2;
#pragma unroll
  for (int i = 0; i < 2; ++i)
#pragma unroll
    for (int j = 0; j < 2; ++j)
#pragma unroll
      for (int r = 0; r < 8; ++r) {
        int gm = EPI_GM(mb, 128, i, r, lane);
        int gn = EPI_GN(nb, 64, j, lane);
        float h = acc[i][j][r] * scale[gm] + bias[gm];
        h = h > 0.f ? h : 0.f;
        size_t o = (size_t)b * Cc * Nn + (size_t)gm * Nn + gn;
        float nv = cur32[o] + h;
        cur32[o] = nv;
        cur16[o] = (_Float16)nv;
        out[((size_t)b * (Ll * Cc) + (size_t)l * Cc + gm) * Nn + gn] = nv;
      }
}

// ---------------------------------------------------------------------------
// Elementwise / reduction kernels
// ---------------------------------------------------------------------------

__global__ void k_f32_to_f16(const float* __restrict__ s,
                             _Float16* __restrict__ d, int n) {
  int i = blockIdx.x * 256 + threadIdx.x;
  if (i < n) d[i] = (_Float16)s[i];
}

// Fused BN params: scale = g/sqrt(v+eps); bias = (bt)*scale + beta - mean*scale
__global__ __launch_bounds__(256) void k_bn_params(
    const float* g1, const float* b1, const float* m1, const float* v1,
    const float* g2, const float* b2, const float* m2, const float* v2,
    const float* gl, const float* bl, const float* ml, const float* vl,
    const float* btl, const float* affine,
    float* scale1, float* bias1, float* scale2, float* bias2,
    float* scaleL, float* biasL, float* betaMean) {
  const int t = threadIdx.x, blk = blockIdx.x;
  constexpr float EPS = 1e-5f;
  if (blk == 0) {
    float s = g1[t] * rsqrtf(v1[t] + EPS);
    scale1[t] = s; bias1[t] = b1[t] - m1[t] * s;
  } else if (blk == 1) {
    float s = g2[t] * rsqrtf(v2[t] + EPS);
    scale2[t] = s; bias2[t] = b2[t] - m2[t] * s;
  } else {
    int l = blk - 2, i = l * Cc + t;
    float s = gl[i] * rsqrtf(vl[i] + EPS);
    scaleL[i] = s;
    biasL[i] = btl[i] * s + bl[i] - ml[i] * s;
    __shared__ float sh[256];
    sh[t] = affine[l * Nn + t];
    __syncthreads();
    for (int k = 128; k > 0; k >>= 1) {
      if (t < k) sh[t] += sh[t + k];
      __syncthreads();
    }
    if (t == 0) betaMean[l] = sh[0] * (1.0f / Nn);
  }
}

// xw[b,c] = softmax_c( max_n xv[b,c,n] )
__global__ __launch_bounds__(256) void k_xw(const _Float16* __restrict__ xv,
                                            float* __restrict__ xw) {
  const int b = blockIdx.x, c = threadIdx.x;
  const _Float16* p = xv + (size_t)b * Cc * Nn + (size_t)c * Nn;
  float mx = -3.0e38f;
  for (int n = 0; n < Nn; n += 8) {
    uint4 u = *(const uint4*)&p[n];
    const _Float16* h = (const _Float16*)&u;
#pragma unroll
    for (int i = 0; i < 8; ++i) mx = fmaxf(mx, (float)h[i]);
  }
  __shared__ float sh[256];
  sh[c] = mx; __syncthreads();
  for (int k = 128; k > 0; k >>= 1) {
    if (c < k) sh[c] = fmaxf(sh[c], sh[c + k]);
    __syncthreads();
  }
  float gmx = sh[0]; __syncthreads();
  float e = __expf(mx - gmx);
  sh[c] = e; __syncthreads();
  for (int k = 128; k > 0; k >>= 1) {
    if (c < k) sh[c] += sh[c + k];
    __syncthreads();
  }
  xw[b * Cc + c] = e / sh[0];
}

// in-place row softmax over m of energy[b,n,:]
__global__ __launch_bounds__(256) void k_softmax_row(float* __restrict__ eng) {
  const int n = blockIdx.x, b = blockIdx.y, m = threadIdx.x;
  float* row = eng + ((size_t)b * Nn + n) * Nn;
  float v = row[m];
  __shared__ float sh[256];
  sh[m] = v; __syncthreads();
  for (int k = 128; k > 0; k >>= 1) {
    if (m < k) sh[m] = fmaxf(sh[m], sh[m + k]);
    __syncthreads();
  }
  float mx = sh[0]; __syncthreads();
  float e = __expf(v - mx);
  sh[m] = e; __syncthreads();
  for (int k = 128; k > 0; k >>= 1) {
    if (m < k) sh[m] += sh[m + k];
    __syncthreads();
  }
  row[m] = e / sh[0];
}

// column renorm: attn16[n,m] = attn[n,m] / (1e-9 + sum_n attn[n,m])
__global__ __launch_bounds__(256) void k_colnorm(const float* __restrict__ attn,
                                                 _Float16* __restrict__ a16) {
  const int b = blockIdx.x, m = threadIdx.x;
  const float* base = attn + (size_t)b * Nn * Nn;
  float s = 0.f;
  for (int n = 0; n < Nn; ++n) s += base[(size_t)n * Nn + m];
  float inv = 1.0f / (1e-9f + s);
  _Float16* o = a16 + (size_t)b * Nn * Nn;
  for (int n = 0; n < Nn; ++n)
    o[(size_t)n * Nn + m] = (_Float16)(base[(size_t)n * Nn + m] * inv);
}

// tsum[b,c] = sum_m xr[b,c,m]
__global__ __launch_bounds__(256) void k_tmean(const float* __restrict__ xr,
                                               float* __restrict__ tsum) {
  const int bc = blockIdx.x, t = threadIdx.x;
  __shared__ float sh[256];
  sh[t] = xr[(size_t)bc * Nn + t];
  __syncthreads();
  for (int k = 128; k > 0; k >>= 1) {
    if (t < k) sh[t] += sh[t + k];
    __syncthreads();
  }
  if (t == 0) tsum[bc] = sh[0];
}

// xs16 = (f16)(cur - t), t = xw * tsum/N + beta_mean
__global__ void k_xs(const float* __restrict__ cur32,
                     const float* __restrict__ xw,
                     const float* __restrict__ tsum,
                     const float* __restrict__ betaMeanL,
                     _Float16* __restrict__ xs16) {
  size_t i = (size_t)blockIdx.x * 256 + threadIdx.x;
  size_t bc = i >> 8;  // [b][c][n], n==256
  float t = xw[bc] * tsum[bc] * (1.0f / Nn) + betaMeanL[0];
  xs16[i] = (_Float16)(cur32[i] - t);
}

// ---------------------------------------------------------------------------
// Host launch
// ---------------------------------------------------------------------------
extern "C" void kernel_launch(void* const* d_in, const int* in_sizes, int n_in,
                              void* d_out, int out_size, void* d_ws, size_t ws_size,
                              hipStream_t stream) {
  (void)in_sizes; (void)n_in; (void)out_size; (void)ws_size;

  constexpr size_t ACT = (size_t)Bb * Cc * Nn;      // 16.7M elements
  char* p = (char*)d_ws;
  auto take = [&](size_t bytes) {
    char* r = p;
    p += (bytes + 255) & ~(size_t)255;
    return r;
  };
  // Aliased regions (lifetimes disjoint):
  _Float16* x16_xs = (_Float16*)take(ACT * 2);      // x16 (convs), then xs16
  float*    eng    = (float*)take(ACT * 4);         // actA32 / energy / x_r
  _Float16* tmp16  = (_Float16*)take(ACT * 2);      // actA16 / attn16
  float*    cur32  = (float*)take(ACT * 4);
  _Float16* cur16  = (_Float16*)take(ACT * 2);
  _Float16* xk16   = (_Float16*)take((size_t)Bb * 64 * Nn * 2);
  _Float16* xv16   = (_Float16*)take(ACT * 2);
  float*    xw     = (float*)take((size_t)Bb * Cc * 4);
  float*    tsum   = (float*)take((size_t)Bb * Cc * 4);
  _Float16* w1     = (_Float16*)take((size_t)Cc * Cc * 2);
  _Float16* w2     = (_Float16*)take((size_t)Cc * Cc * 2);
  _Float16* wqk    = (_Float16*)take((size_t)Ll * 64 * Cc * 2);
  _Float16* wv     = (_Float16*)take((size_t)Ll * Cc * Cc * 2);
  _Float16* wt     = (_Float16*)take((size_t)Ll * Cc * Cc * 2);
  float* scale1 = (float*)take(Cc * 4);
  float* bias1  = (float*)take(Cc * 4);
  float* scale2 = (float*)take(Cc * 4);
  float* bias2  = (float*)take(Cc * 4);
  float* scaleL = (float*)take(Ll * Cc * 4);
  float* biasL  = (float*)take(Ll * Cc * 4);
  float* betaMean = (float*)take(256);

  float* out = (float*)d_out;

  auto cvt = [&](const void* s, _Float16* d, int n) {
    k_f32_to_f16<<<(n + 255) / 256, 256, 0, stream>>>((const float*)s, d, n);
  };
  cvt(d_in[0], x16_xs, (int)ACT);
  cvt(d_in[1], w1, Cc * Cc);
  cvt(d_in[2], w2, Cc * Cc);
  cvt(d_in[11], wqk, Ll * 64 * Cc);
  cvt(d_in[12], wv, Ll * Cc * Cc);
  cvt(d_in[14], wt, Ll * Cc * Cc);

  k_bn_params<<<2 + Ll, 256, 0, stream>>>(
      (const float*)d_in[3], (const float*)d_in[4], (const float*)d_in[5], (const float*)d_in[6],
      (const float*)d_in[7], (const float*)d_in[8], (const float*)d_in[9], (const float*)d_in[10],
      (const float*)d_in[16], (const float*)d_in[17], (const float*)d_in[18], (const float*)d_in[19],
      (const float*)d_in[15], (const float*)d_in[20],
      scale1, bias1, scale2, bias2, scaleL, biasL, betaMean);

  const dim3 gM256(Nn / 64, Cc / 128, Bb);  // 128x64 tiles

  // conv1: x16 -> (eng, tmp16) ; conv2: tmp16 -> (cur32, cur16)
  k_gemm_convbn<<<gM256, 256, 0, stream>>>(w1, x16_xs, scale1, bias1, eng, tmp16);
  k_gemm_convbn<<<gM256, 256, 0, stream>>>(w2, tmp16, scale2, bias2, cur32, cur16);

  for (int l = 0; l < Ll; ++l) {
    const _Float16* wqk_l = wqk + (size_t)l * 64 * Cc;
    const _Float16* wv_l  = wv + (size_t)l * Cc * Cc;
    const _Float16* wt_l  = wt + (size_t)l * Cc * Cc;
    const float* bv_l = (const float*)d_in[13] + (size_t)l * Cc;

    k_gemm_xk<<<dim3(Nn / 128, 1, Bb), 256, 0, stream>>>(wqk_l, cur16, xk16);
    k_gemm_xv<<<gM256, 256, 0, stream>>>(wv_l, cur16, bv_l, xv16);
    k_xw<<<Bb, 256, 0, stream>>>(xv16, xw);
    k_gemm_energy<<<dim3(Nn / 64, Nn / 128, Bb), 256, 0, stream>>>(xk16, eng);
    k_softmax_row<<<dim3(Nn, Bb), 256, 0, stream>>>(eng);
    k_colnorm<<<Bb, 256, 0, stream>>>(eng, tmp16);          // attn16 in tmp16
    k_gemm_xr<<<gM256, 256, 0, stream>>>(xv16, tmp16, eng); // raw x_r reuses eng
    k_tmean<<<Bb * Cc, 256, 0, stream>>>(eng, tsum);
    k_xs<<<(unsigned)(ACT / 256), 256, 0, stream>>>(cur32, xw, tsum,
                                                    betaMean + l, x16_xs);
    k_gemm_out<<<gM256, 256, 0, stream>>>(wt_l, x16_xs, scaleL + (size_t)l * Cc,
                                          biasL + (size_t)l * Cc, cur32, cur16,
                                          out, l);
  }
}
